// GMPAHead_16484084483646
// MI455X (gfx1250) — compile-verified, matrix operand order checked by reading, not run on hardware
//
#include <hip/hip_runtime.h>
#include <hip/hip_bf16.h>
#include <stdint.h>

typedef __attribute__((ext_vector_type(16))) __bf16 v16bf;
typedef __attribute__((ext_vector_type(8)))  float  v8f;
typedef __attribute__((ext_vector_type(4)))  float  f32x4;
typedef __attribute__((ext_vector_type(4)))  unsigned int u32x4;
typedef __attribute__((ext_vector_type(8)))  int i32x8;
typedef __attribute__((ext_vector_type(4)))  int i32x4;

#define NBINS 31
#define BB    128
#define CC    256
#define TT    30
#define HH    16
#define WW    11
#define RED   16
#define OUTC  256
#define NCLS  10000
#define LEAKY 0.01f

#if defined(__has_builtin)
#if __has_builtin(__builtin_amdgcn_tensor_load_to_lds)
#define USE_TDM 1
#endif
#endif
#ifndef USE_TDM
#define USE_TDM 0
#endif

#if USE_TDM
#if __clang_major__ >= 23
#define TDM_LOAD(g0, g1, za, zb)                                     \
  do { i32x8 _z8 = {0,0,0,0,0,0,0,0};                                \
       __builtin_amdgcn_tensor_load_to_lds(g0, g1, za, zb, _z8, 0);  \
  } while (0)
#else
#define TDM_LOAD(g0, g1, za, zb)                                     \
  __builtin_amdgcn_tensor_load_to_lds(g0, g1, za, zb, 0)
#endif
#if __has_builtin(__builtin_amdgcn_s_wait_tensorcnt)
#define WAIT_TENSOR() __builtin_amdgcn_s_wait_tensorcnt(0)
#else
#define WAIT_TENSOR() asm volatile("s_wait_tensorcnt 0x0" ::: "memory")
#endif
#endif

// ---------------------------------------------------------------------------
// K1: horizontal strip pooling. Block = 16 channels x 16 rows. For each (b,t,c)
// compute per-row (sum,max) over W=11, then hierarchically combine rows into
// the 31 GaitSet bins (p=1,2,4,8,16 -> bin bases 0,1,3,7,15).
// P layout: [n][B][T][C].  x is streamed once -> non-temporal loads.
// ---------------------------------------------------------------------------
__global__ void __launch_bounds__(256) k_hmap(const float* __restrict__ x,
                                              float* __restrict__ P) {
  int tid = threadIdx.x;
  int cl  = tid >> 4;
  int h   = tid & 15;
  int blk = blockIdx.x;
  int cblk = blk & 15;
  int t    = (blk >> 4) % TT;
  int b    = (blk >> 4) / TT;
  int c    = cblk * 16 + cl;

  const float* row = x + ((((size_t)b*CC + c)*TT + t)*HH + h)*WW;
  float s = 0.f, m = -3.402823e38f;
#pragma unroll
  for (int w = 0; w < WW; ++w) {
    float v = __builtin_nontemporal_load(row + w);
    s += v; m = fmaxf(m, v);
  }

  __shared__ float S[16][16];
  __shared__ float M[16][16];
  S[cl][h] = s; M[cl][h] = m;
  __syncthreads();

  size_t baseIdx = (((size_t)b*TT) + t)*CC + c;
  size_t plane   = (size_t)BB*TT*CC;

  P[(size_t)(15 + h)*plane + baseIdx] = S[cl][h] * (1.f/(1.f*WW)) + M[cl][h];
  if (h < 8) {
    float ss = S[cl][2*h] + S[cl][2*h+1];
    float mm = fmaxf(M[cl][2*h], M[cl][2*h+1]);
    P[(size_t)(7 + h)*plane + baseIdx] = ss * (1.f/(2.f*WW)) + mm;
  }
  if (h < 4) {
    float ss = 0.f, mm = -3.402823e38f;
#pragma unroll
    for (int i = 0; i < 4; ++i) { ss += S[cl][4*h+i]; mm = fmaxf(mm, M[cl][4*h+i]); }
    P[(size_t)(3 + h)*plane + baseIdx] = ss * (1.f/(4.f*WW)) + mm;
  }
  if (h < 2) {
    float ss = 0.f, mm = -3.402823e38f;
#pragma unroll
    for (int i = 0; i < 8; ++i) { ss += S[cl][8*h+i]; mm = fmaxf(mm, M[cl][8*h+i]); }
    P[(size_t)(1 + h)*plane + baseIdx] = ss * (1.f/(8.f*WW)) + mm;
  }
  if (h == 0) {
    float ss = 0.f, mm = -3.402823e38f;
#pragma unroll
    for (int i = 0; i < 16; ++i) { ss += S[cl][i]; mm = fmaxf(mm, M[cl][i]); }
    P[baseIdx] = ss * (1.f/(16.f*WW)) + mm;
  }
}

// ---------------------------------------------------------------------------
// K2: Pmax[n][b][c] = max_t P[n][b][t][c]
// ---------------------------------------------------------------------------
__global__ void __launch_bounds__(256) k_pmax(const float* __restrict__ P,
                                              float* __restrict__ Pmax) {
  int n = blockIdx.x / BB, b = blockIdx.x % BB, c = threadIdx.x;
  const float* p = P + (((size_t)n*BB + b)*TT)*CC + c;
  float m = -3.402823e38f;
  for (int t = 0; t < TT; ++t) m = fmaxf(m, p[(size_t)t*CC]);
  Pmax[((size_t)n*BB + b)*CC + c] = m;
}

// ---------------------------------------------------------------------------
// K3: fused mask GEMM + softmax-over-T + attention pooling. Block per (n,b).
// ---------------------------------------------------------------------------
__global__ void __launch_bounds__(256) k_maskpool(const float* __restrict__ P,
                                                  const float* __restrict__ Wmask,
                                                  float* __restrict__ pooled) {
  __shared__ float Pl[TT*CC];
  __shared__ float Wl[CC*RED];
  __shared__ float Ml[TT*RED];
  int n = blockIdx.x / BB, b = blockIdx.x % BB;
  int tid = threadIdx.x;

  const float* Pg = P + ((size_t)n*BB + b)*TT*CC;
  for (int i = tid; i < TT*CC; i += 256) Pl[i] = Pg[i];
  const float* Wg = Wmask + (size_t)n*CC*RED;
  for (int i = tid; i < CC*RED; i += 256) Wl[i] = Wg[i];
  __syncthreads();

  for (int e = tid; e < TT*RED; e += 256) {
    int t = e / RED, r = e % RED;
    float acc = 0.f;
    for (int c = 0; c < CC; ++c) acc += Pl[t*CC + c] * Wl[c*RED + r];
    Ml[e] = acc;
  }
  __syncthreads();

  if (tid < RED) {
    int r = tid;
    float mx = -3.402823e38f;
    for (int t = 0; t < TT; ++t) mx = fmaxf(mx, Ml[t*RED + r]);
    float s = 0.f;
    for (int t = 0; t < TT; ++t) { float e = __expf(Ml[t*RED + r] - mx); Ml[t*RED + r] = e; s += e; }
    float inv = 1.f / s;
    for (int t = 0; t < TT; ++t) Ml[t*RED + r] *= inv;
  }
  __syncthreads();

  int c = tid;
  float acc[RED];
#pragma unroll
  for (int r = 0; r < RED; ++r) acc[r] = 0.f;
  for (int t = 0; t < TT; ++t) {
    float pv = Pl[t*CC + c];
#pragma unroll
    for (int r = 0; r < RED; ++r) acc[r] += pv * Ml[t*RED + r];
  }
  float* pg = pooled + ((size_t)n*BB + b)*(CC*RED) + (size_t)c*RED;
#pragma unroll
  for (int r = 0; r < RED; ++r) pg[r] = acc[r];
}

// ---------------------------------------------------------------------------
// bf16 WMMA GEMM: per-bin [128 x Ncols] = A[128 x Kdim] @ B[Kdim x Ncols].
// 8 waves x 16 M-rows, 64-wide N tile, K-step 32, double-buffered LDS.
// Staging issues all global b128 loads into independent registers first (one
// loadcnt wait per tile, loads overlap the WMMAs of the previous K-step),
// then converts + stores to LDS.  B matrices are streamed once -> TH=NT.
// For bf16 A (MODE 1/2) the 128x32 tile is DMA'd by the TDM when available.
// MODE 0: A=pooled(f32), B=Wout;      epi: LeakyReLU + Pmax -> feat bf16
// MODE 1: A=feat(bf16),  B=FForward;  epi: outputs f32 at [b][n][o]
// MODE 2: A=bnn(bf16),   B=fc1d;      epi: * inv_colnorm -> logits [b][n][k]
// ---------------------------------------------------------------------------
template<int MODE>
__global__ void __launch_bounds__(256) k_wmma_gemm(
    const void* __restrict__ Abase, const float* __restrict__ Bbase,
    float* __restrict__ OutF, __bf16* __restrict__ OutB,
    const float* __restrict__ Aux,
    int Kdim, int Ncols, int nTilesN)
{
  __shared__ alignas(16) __bf16 Asub[2][128*32];
  __shared__ alignas(16) __bf16 Bsub[2][64*32];   // transposed: [ncol][k]

  int n  = blockIdx.x / nTilesN;
  int n0 = (blockIdx.x % nTilesN) * 64;
  int tid = threadIdx.x;

  const float*  Af32 = (const float*) Abase + (size_t)n*128*Kdim;
  const __bf16* Ab16 = (const __bf16*)Abase + (size_t)n*128*Kdim;
  const float*  Bp   = Bbase + (size_t)n*Kdim*Ncols;

  int lane = tid & 31, wv = tid >> 5;
  int mb = wv * 16;
  int m  = lane & 15, hi = lane >> 4, nl = lane & 15;

  v8f zero = {0.f,0.f,0.f,0.f,0.f,0.f,0.f,0.f};
  v8f acc[4];
#pragma unroll
  for (int i = 0; i < 4; ++i) acc[i] = zero;

  auto stage = [&](int buf, int k0) {
    // ================= phase 1: issue ALL global loads =================
    f32x4 fa[4];
    uint4 ua[2];
    int r0 = tid >> 3;             // 0..31
    int c4 = (tid & 7) << 2;       // 0,4,...,28
    if (MODE == 0) {
#pragma unroll
      for (int i = 0; i < 4; ++i)
        fa[i] = __builtin_nontemporal_load(
            (const f32x4*)(Af32 + (size_t)(r0 + i*32)*Kdim + k0 + c4));
    } else {
#if USE_TDM
      if (wv == 0) {
        // Tensor DMA: 2D tile 32 x 128 (dim0 x dim1), 2-byte elements,
        // tensor row stride = Kdim, dest = Asub[buf] (contiguous 8KB).
        uint64_t ga = (uint64_t)(uintptr_t)(Ab16 + k0);
        u32x4 g0;
        g0[0] = 1u;                                            // count=1
        g0[1] = (unsigned)(uintptr_t)&Asub[buf][0];            // lds_addr
        g0[2] = (unsigned)(ga & 0xffffffffu);                  // global_addr lo
        g0[3] = (unsigned)((ga >> 32) & 0x1ffffffu) | 0x80000000u; // hi | type=2
        i32x8 g1;
        g1[0] = 0x10000;                                       // data_size=2B
        g1[1] = (int)(((unsigned)Kdim & 0xffffu) << 16);       // tensor_dim0 lo
        g1[2] = (int)((((unsigned)Kdim >> 16) & 0xffffu) | (128u << 16));
        g1[3] = (int)(32u << 16);                              // tile_dim0=32
        g1[4] = 128;                                           // tile_dim1=128
        g1[5] = Kdim;                                          // dim0_stride lo
        g1[6] = 0;
        g1[7] = 0;
        i32x4 z4 = {0,0,0,0};
        TDM_LOAD(g0, g1, z4, z4);
      }
#else
#pragma unroll
      for (int i = 0; i < 2; ++i) {        // 512 uint4 = 128 rows x 4 uint4
        int e = tid + i*256;
        int rr = e >> 2, qq = (e & 3) << 3;
        ua[i] = *(const uint4*)(Ab16 + (size_t)rr*Kdim + k0 + qq);
      }
#endif
    }
    int kp  = tid >> 4;            // 0..15 -> k = 2*kp
    int nc4 = (tid & 15) << 2;     // 0..60
    int gk  = k0 + kp*2;
    int gc  = n0 + nc4;
    const float* p0 = Bp + (size_t)gk*Ncols;
    const float* p1 = p0 + Ncols;
    f32x4 rb0 = {0.f,0.f,0.f,0.f}, rb1 = {0.f,0.f,0.f,0.f};
    if (gc + 4 <= Ncols) {
      rb0 = __builtin_nontemporal_load((const f32x4*)(p0 + gc));
      rb1 = __builtin_nontemporal_load((const f32x4*)(p1 + gc));
    } else {
#pragma unroll
      for (int j = 0; j < 4; ++j) {
        rb0[j] = (gc + j < Ncols) ? p0[gc + j] : 0.f;
        rb1[j] = (gc + j < Ncols) ? p1[gc + j] : 0.f;
      }
    }
    // ================= phase 2: convert + LDS stores =================
    if (MODE == 0) {
#pragma unroll
      for (int i = 0; i < 4; ++i) {
        union { __bf16 h[4]; uint2 u; } pk;
        pk.h[0] = (__bf16)fa[i][0]; pk.h[1] = (__bf16)fa[i][1];
        pk.h[2] = (__bf16)fa[i][2]; pk.h[3] = (__bf16)fa[i][3];
        *(uint2*)&Asub[buf][(r0 + i*32)*32 + c4] = pk.u;
      }
    } else {
#if !USE_TDM
#pragma unroll
      for (int i = 0; i < 2; ++i) {
        int e = tid + i*256;
        int rr = e >> 2, qq = (e & 3) << 3;
        *(uint4*)&Asub[buf][rr*32 + qq] = ua[i];
      }
#endif
    }
#pragma unroll
    for (int j = 0; j < 4; ++j) {
      union { __bf16 h[2]; uint32_t u; } q;
      q.h[0] = (__bf16)rb0[j]; q.h[1] = (__bf16)rb1[j];
      *(uint32_t*)&Bsub[buf][(nc4 + j)*32 + 2*kp] = q.u;
    }
  };

  int nsteps = Kdim >> 5;
  stage(0, 0);
  for (int it = 0; it < nsteps; ++it) {
#if USE_TDM
    if (MODE != 0 && wv == 0) WAIT_TENSOR();
#endif
    __syncthreads();
    if (it + 1 < nsteps) stage((it + 1) & 1, (it + 1) << 5);

    int buf = it & 1;
    union { uint32_t u[8]; v16bf v; } afr;
#pragma unroll
    for (int v = 0; v < 8; ++v) {
      int kk = (v < 4) ? (2*v + 8*hi) : (16 + 2*(v-4) + 8*hi);
      afr.u[v] = *(const uint32_t*)&Asub[buf][(mb + m)*32 + kk];
    }
#pragma unroll
    for (int c16 = 0; c16 < 4; ++c16) {
      union { uint32_t u[8]; v16bf v; } bfr;
#pragma unroll
      for (int v = 0; v < 8; ++v) {
        int kk = (v < 4) ? (2*v + 8*hi) : (16 + 2*(v-4) + 8*hi);
        bfr.u[v] = *(const uint32_t*)&Bsub[buf][(c16*16 + nl)*32 + kk];
      }
      acc[c16] = __builtin_amdgcn_wmma_f32_16x16x32_bf16(
          false, afr.v, false, bfr.v, (short)0, acc[c16], false, false);
    }
  }

#pragma unroll
  for (int c16 = 0; c16 < 4; ++c16) {
#pragma unroll
    for (int j = 0; j < 8; ++j) {
      int row = mb + j + 8*hi;
      int col = n0 + c16*16 + nl;
      float v = acc[c16][j];
      if (MODE == 0) {
        v = (v >= 0.f) ? v : LEAKY * v;
        v += Aux[((size_t)n*128 + row)*256 + col];
        OutB[((size_t)n*128 + row)*256 + col] = (__bf16)v;
      } else if (MODE == 1) {
        OutF[((size_t)row*NBINS + n)*256 + col] = v;
      } else {
        if (col < Ncols)
          __builtin_nontemporal_store(
              v * Aux[(size_t)n*NCLS + col],
              &OutF[((size_t)row*NBINS + n)*(size_t)NCLS + col]);
      }
    }
  }
}

// ---------------------------------------------------------------------------
__global__ void __launch_bounds__(256) k_bnstats(const float* __restrict__ out1,
    const float* __restrict__ gamma, const float* __restrict__ beta,
    float* __restrict__ scale, float* __restrict__ bias) {
  int n = blockIdx.x, o = threadIdx.x;
  float s = 0.f, s2 = 0.f;
  for (int b = 0; b < BB; ++b) {
    float v = out1[((size_t)b*NBINS + n)*OUTC + o];
    s += v; s2 += v*v;
  }
  float mean = s * (1.f/BB);
  float var  = s2 * (1.f/BB) - mean*mean;
  float sc = gamma[n*OUTC + o] * rsqrtf(var + 1e-5f);
  scale[n*OUTC + o] = sc;
  bias[n*OUTC + o]  = beta[n*OUTC + o] - mean*sc;
}

__global__ void __launch_bounds__(256) k_bnn(const float* __restrict__ out1,
    const float* __restrict__ scale, const float* __restrict__ bias,
    __bf16* __restrict__ bnn) {
  __shared__ float red[256];
  int n = blockIdx.x / BB, b = blockIdx.x % BB, o = threadIdx.x;
  float v = out1[((size_t)b*NBINS + n)*OUTC + o] * scale[n*OUTC + o] + bias[n*OUTC + o];
  red[o] = v*v;
  __syncthreads();
  for (int s = 128; s > 0; s >>= 1) {
    if (o < s) red[o] += red[o + s];
    __syncthreads();
  }
  float nm = fmaxf(sqrtf(red[0]), 1e-12f);
  bnn[((size_t)n*BB + b)*OUTC + o] = (__bf16)(v / nm);
}

__global__ void __launch_bounds__(256) k_fcnorm(const float* __restrict__ fc,
                                                float* __restrict__ inv) {
  int n = blockIdx.x / 40;
  int k = (blockIdx.x % 40)*256 + threadIdx.x;
  if (k >= NCLS) return;
  const float* col = fc + (size_t)n*OUTC*NCLS + k;
  float s = 0.f;
  for (int o = 0; o < OUTC; ++o) { float v = col[(size_t)o*NCLS]; s += v*v; }
  inv[(size_t)n*NCLS + k] = 1.f / fmaxf(sqrtf(s), 1e-12f);
}

// ---------------------------------------------------------------------------
extern "C" void kernel_launch(void* const* d_in, const int* in_sizes, int n_in,
                              void* d_out, int out_size, void* d_ws, size_t ws_size,
                              hipStream_t stream) {
  const float* x     = (const float*)d_in[0];
  const float* Wmask = (const float*)d_in[1];
  const float* Wout  = (const float*)d_in[2];
  const float* FFw   = (const float*)d_in[3];
  const float* gamma = (const float*)d_in[4];
  const float* beta  = (const float*)d_in[5];
  const float* fc1d  = (const float*)d_in[6];

  float* out1 = (float*)d_out;                         // [B,31,256]
  float* out2 = out1 + (size_t)BB*NBINS*OUTC;          // [B,31,10000]

  char* w = (char*)d_ws;
  size_t nP    = (size_t)NBINS*BB*TT*CC;
  size_t nPmax = (size_t)NBINS*BB*CC;
  size_t nPool = (size_t)NBINS*BB*CC*RED;

  float*  P       = (float*) w;  w += nP*4;
  float*  Pmax    = (float*) w;  w += nPmax*4;
  float*  pooled  = (float*) w;  w += nPool*4;
  __bf16* feat    = (__bf16*)w;  w += nPmax*2;
  float*  bnscale = (float*) w;  w += (size_t)NBINS*OUTC*4;
  float*  bnbias  = (float*) w;  w += (size_t)NBINS*OUTC*4;
  __bf16* bnn     = (__bf16*)w;  w += nPmax*2;
  float*  invfcn  = (float*) w;  w += (size_t)NBINS*NCLS*4;

  k_hmap    <<<BB*TT*(CC/16), 256, 0, stream>>>(x, P);
  k_pmax    <<<NBINS*BB,      256, 0, stream>>>(P, Pmax);
  k_maskpool<<<NBINS*BB,      256, 0, stream>>>(P, Wmask, pooled);

  k_wmma_gemm<0><<<NBINS*4,   256, 0, stream>>>(pooled, Wout, nullptr, feat, Pmax,
                                                CC*RED, CC, 4);
  k_wmma_gemm<1><<<NBINS*4,   256, 0, stream>>>(feat, FFw, out1, nullptr, nullptr,
                                                CC, OUTC, 4);
  k_bnstats <<<NBINS,         256, 0, stream>>>(out1, gamma, beta, bnscale, bnbias);
  k_bnn     <<<NBINS*BB,      256, 0, stream>>>(out1, bnscale, bnbias, bnn);
  k_fcnorm  <<<NBINS*40,      256, 0, stream>>>(fc1d, invfcn);
  k_wmma_gemm<2><<<NBINS*157, 256, 0, stream>>>(bnn, fc1d, out2, nullptr, invfcn,
                                                OUTC, NCLS, 157);
}
